// InferTree_16226386444598
// MI455X (gfx1250) — compile-verified, matrix operand order checked by reading, not run on hardware
//
#include <hip/hip_runtime.h>
#include <hip/hip_bf16.h>
#include <stdint.h>

#define BATCH   4096
#define DIM     2048
#define NCOLS   1221
#define NPAD    1280
#define OUTC    1001
#define LDSW    40      // halves per LDS row (8-half pad -> 80B rows, 16B aligned)

typedef __bf16 bf16_t;
typedef bf16_t v16bf __attribute__((ext_vector_type(16)));
typedef bf16_t v8bf  __attribute__((ext_vector_type(8)));
typedef float  v8f   __attribute__((ext_vector_type(8)));

__device__ __forceinline__ void split_f32(float f, bf16_t& hi, bf16_t& lo) {
    hi = (bf16_t)f;
    lo = (bf16_t)(f - (float)hi);
}

// ---- CDNA5 async global->LDS copy (ASYNCcnt-tracked), GV addressing mode.
__device__ __forceinline__ void async_b128(uint32_t ldsByteOff, const bf16_t* g) {
    asm volatile("global_load_async_to_lds_b128 %0, %1, off"
                 :: "v"(ldsByteOff), "v"(g) : "memory");
}
__device__ __forceinline__ void wait_async0() {
    asm volatile("s_wait_asynccnt 0" ::: "memory");
}

// Load a 16x32 bf16 fragment (A layout; B identical since W is N-major) from an
// LDS plane with LDSW-half row stride.
// lanes 0-15: row = base+lane,    VGPR0-3 = K0..7,  VGPR4-7 = K16..23
// lanes16-31: row = base+lane-16, VGPR0-3 = K8..15, VGPR4-7 = K24..31
__device__ __forceinline__ v16bf load_frag(const bf16_t* plane, int rowBase, int lane) {
    int row = rowBase + (lane & 15);
    int kh  = lane >> 4;
    const bf16_t* p = plane + row * LDSW + kh * 8;
    v8bf f0 = *(const v8bf*)(p);               // ds_load_b128
    v8bf f1 = *(const v8bf*)(p + 16);          // ds_load_b128
    return __builtin_shufflevector(f0, f1, 0,1,2,3,4,5,6,7,8,9,10,11,12,13,14,15);
}

// ---------------------------------------------------------------- prep kernels
__global__ __launch_bounds__(256)
void hs_prep_x(const float* __restrict__ x,
               bf16_t* __restrict__ xHi, bf16_t* __restrict__ xLo) {
    int idx = blockIdx.x * 256 + threadIdx.x;  // exact grid: BATCH*DIM/256
    bf16_t h, l; split_f32(x[idx], h, l);
    xHi[idx] = h; xLo[idx] = l;
}

__global__ __launch_bounds__(256)
void hs_prep_w(const float* __restrict__ W1, const float* __restrict__ W2,
               const float* __restrict__ W3,
               bf16_t* __restrict__ wHi, bf16_t* __restrict__ wLo) {
    int idx = blockIdx.x * 256 + threadIdx.x;
    if (idx >= NPAD * DIM) return;
    int col = idx / DIM;
    float w = 0.f;
    if (col < 11)          w = W1[idx];
    else if (col < 121)    w = W2[idx - 11 * DIM];
    else if (col < NCOLS)  w = W3[idx - 121 * DIM];
    bf16_t h, l; split_f32(w, h, l);
    wHi[idx] = h; wLo[idx] = l;
}

__global__ __launch_bounds__(256)
void hs_prep_bias(const float* __restrict__ b1, const float* __restrict__ b2,
                  const float* __restrict__ b3, float* __restrict__ bias) {
    int col = blockIdx.x * 256 + threadIdx.x;
    if (col >= NPAD) return;
    float v = 0.f;
    if (col < 11)          v = b1[col];
    else if (col < 121)    v = b2[col - 11];
    else if (col < NCOLS)  v = b3[col - 121];
    bias[col] = v;
}

// ---------------------------------------------------------------- GEMM kernel
// logits[4096,1280] = x * W^T; fp32 via bf16 hi/lo split (3 WMMAs / 16x16x32).
// Double-buffered LDS fed by global_load_async_to_lds_b128.
__global__ __launch_bounds__(256)
void hs_gemm(const bf16_t* __restrict__ xHi, const bf16_t* __restrict__ xLo,
             const bf16_t* __restrict__ wHi, const bf16_t* __restrict__ wLo,
             const float* __restrict__ bias, float* __restrict__ logits) {
    __shared__ bf16_t sAhi[2][128 * LDSW];
    __shared__ bf16_t sAlo[2][128 * LDSW];
    __shared__ bf16_t sBhi[2][ 64 * LDSW];
    __shared__ bf16_t sBlo[2][ 64 * LDSW];

    const int tid  = threadIdx.x;
    const int lane = tid & 31;
    const int wave = tid >> 5;
    const int wm   = wave & 3;                 // 4 waves along M -> 32-row slabs
    const int wn   = wave >> 2;                // 2 waves along N -> 32-col slabs
    const int rowBase = blockIdx.y * 128;
    const int colBase = blockIdx.x * 64;

    const uint32_t aHiB = (uint32_t)(size_t)&sAhi[0][0];
    const uint32_t aLoB = (uint32_t)(size_t)&sAlo[0][0];
    const uint32_t bHiB = (uint32_t)(size_t)&sBhi[0][0];
    const uint32_t bLoB = (uint32_t)(size_t)&sBlo[0][0];
    const uint32_t bufA = 128 * LDSW * 2;      // bytes per A buffer
    const uint32_t bufB =  64 * LDSW * 2;      // bytes per B buffer

    // Issue all async copies for one K-step into buffer `buf`.
    auto issue = [&](int buf, int k0) {
        #pragma unroll
        for (int t = 0; t < 2; ++t) {          // A: 128 rows x 4 x b128
            int s = tid + t * 256;
            int row = s >> 2, q = s & 3;
            size_t   g = (size_t)(rowBase + row) * DIM + k0 + q * 8;
            uint32_t l = (uint32_t)(row * (LDSW * 2) + q * 16) + (uint32_t)buf * bufA;
            async_b128(aHiB + l, xHi + g);
            async_b128(aLoB + l, xLo + g);
        }
        {                                       // B: 64 rows x 4 x b128
            int row = tid >> 2, q = tid & 3;
            size_t   g = (size_t)(colBase + row) * DIM + k0 + q * 8;
            uint32_t l = (uint32_t)(row * (LDSW * 2) + q * 16) + (uint32_t)buf * bufB;
            async_b128(bHiB + l, wHi + g);
            async_b128(bLoB + l, wLo + g);
        }
    };

    v8f acc[2][2] = {};

    issue(0, 0);
    wait_async0();
    __syncthreads();

    int cur = 0;
    for (int k0 = 0; k0 < DIM; k0 += 32) {
        int nxt = cur ^ 1;
        if (k0 + 32 < DIM) issue(nxt, k0 + 32);    // overlap copies with math

        v16bf aH[2], aL[2], bH[2], bL[2];
        #pragma unroll
        for (int tm = 0; tm < 2; ++tm) {
            aH[tm] = load_frag(&sAhi[cur][0], wm * 32 + tm * 16, lane);
            aL[tm] = load_frag(&sAlo[cur][0], wm * 32 + tm * 16, lane);
        }
        #pragma unroll
        for (int tn = 0; tn < 2; ++tn) {
            bH[tn] = load_frag(&sBhi[cur][0], wn * 32 + tn * 16, lane);
            bL[tn] = load_frag(&sBlo[cur][0], wn * 32 + tn * 16, lane);
        }
        #pragma unroll
        for (int tm = 0; tm < 2; ++tm)
            #pragma unroll
            for (int tn = 0; tn < 2; ++tn) {
                v8f c = acc[tm][tn];
                c = __builtin_amdgcn_wmma_f32_16x16x32_bf16(false, aL[tm], false, bH[tn],
                                                            (short)0, c, false, false);
                c = __builtin_amdgcn_wmma_f32_16x16x32_bf16(false, aH[tm], false, bL[tn],
                                                            (short)0, c, false, false);
                c = __builtin_amdgcn_wmma_f32_16x16x32_bf16(false, aH[tm], false, bH[tn],
                                                            (short)0, c, false, false);
                acc[tm][tn] = c;
            }

        wait_async0();        // own next-buffer copies landed
        __syncthreads();      // all waves: copies landed & cur fully consumed
        cur = nxt;
    }

    // ---- epilogue: +bias, store fp32 logits.
    // C/D layout: VGPR r -> M=r (lanes 0-15) / M=r+8 (lanes 16-31); N = lane%16.
    #pragma unroll
    for (int tm = 0; tm < 2; ++tm)
        #pragma unroll
        for (int tn = 0; tn < 2; ++tn) {
            int m0 = rowBase + wm * 32 + tm * 16 + ((lane >> 4) << 3);
            int n0 = colBase + wn * 32 + tn * 16 + (lane & 15);
            float bb = bias[n0];
            #pragma unroll
            for (int r = 0; r < 8; ++r)
                logits[(size_t)(m0 + r) * NPAD + n0] = acc[tm][tn][r] + bb;
        }
}

// ---------------------------------------------------------------- output fill
__global__ __launch_bounds__(256)
void hs_out(const float* __restrict__ logits, float* __restrict__ out) {
    int e = blockIdx.x * 256 + threadIdx.x;
    const int total = BATCH * OUTC;
    if (e > total) return;
    if (e == total) { out[total] = 0.f; return; }   // zero penalty accumulator
    int b = e / OUTC, c = e % OUTC;
    float v = 0.f;
    if (c > 0) {
        int m = c - 1;
        v = logits[(size_t)b * NPAD + 121 + (m / 10) * 11 + 1 + (m % 10)];
    }
    out[e] = v;
}

// ---------------------------------------------------------------- penalty
__global__ __launch_bounds__(256)
void hs_penalty(const float* __restrict__ logits, const int* __restrict__ labels,
                float* __restrict__ penalty) {
    __shared__ float red[256];
    int idx = blockIdx.x * 256 + threadIdx.x;        // exact grid: 4096*111
    int b = idx / 111, n = idx % 111;
    int lab = labels[b];
    int t; float w;
    if (n == 0)      { t = lab / 100 + 1;                                      w = 1.f; }
    else if (n < 11) { int nd = n - 1;  t = (lab / 100 == nd) ? ((lab / 10) % 10 + 1) : 0; w = 0.5f; }
    else             { int nd = n - 11; t = (lab / 10  == nd) ? (lab % 10 + 1) : 0;        w = 1.f / 3.f; }
    const float* l = logits + (size_t)b * NPAD + n * 11;
    float mx = l[0];
    #pragma unroll
    for (int c = 1; c < 11; ++c) mx = fmaxf(mx, l[c]);
    float s = 0.f;
    #pragma unroll
    for (int c = 0; c < 11; ++c) s += __expf(l[c] - mx);
    float ce = __logf(s) + mx - l[t];
    red[threadIdx.x] = w * ce * (1.0f / BATCH);
    __syncthreads();
    for (int off = 128; off > 0; off >>= 1) {
        if (threadIdx.x < off) red[threadIdx.x] += red[threadIdx.x + off];
        __syncthreads();
    }
    if (threadIdx.x == 0) atomicAdd(penalty, red[0]);
}

// ---------------------------------------------------------------- launcher
extern "C" void kernel_launch(void* const* d_in, const int* in_sizes, int n_in,
                              void* d_out, int out_size, void* d_ws, size_t ws_size,
                              hipStream_t stream) {
    (void)in_sizes; (void)n_in; (void)out_size; (void)ws_size;
    const float* x      = (const float*)d_in[0];
    const int*   labels = (const int*)  d_in[1];
    const float* W1     = (const float*)d_in[2];
    const float* b1     = (const float*)d_in[3];
    const float* W2     = (const float*)d_in[4];
    const float* b2     = (const float*)d_in[5];
    const float* W3     = (const float*)d_in[6];
    const float* b3     = (const float*)d_in[7];
    float* out = (float*)d_out;

    // ws layout (bytes):
    //   xHi 16M | xLo 16M | wHi 5.25M | wLo 5.25M | bias 5K | logits 21M  (~62MB)
    char* ws = (char*)d_ws;
    const size_t xPlane = (size_t)BATCH * DIM * 2;
    const size_t wPlane = (size_t)NPAD * DIM * 2;
    bf16_t* xHi    = (bf16_t*)(ws);
    bf16_t* xLo    = (bf16_t*)(ws + xPlane);
    bf16_t* wHi    = (bf16_t*)(ws + 2 * xPlane);
    bf16_t* wLo    = (bf16_t*)(ws + 2 * xPlane + wPlane);
    float*  bias   = (float*) (ws + 2 * xPlane + 2 * wPlane);
    float*  logits = (float*) (ws + 2 * xPlane + 2 * wPlane + (size_t)NPAD * 4);

    hs_prep_x   <<<BATCH * DIM / 256,          256, 0, stream>>>(x, xHi, xLo);
    hs_prep_w   <<<(NPAD * DIM + 255) / 256,   256, 0, stream>>>(W1, W2, W3, wHi, wLo);
    hs_prep_bias<<<(NPAD + 255) / 256,         256, 0, stream>>>(b1, b2, b3, bias);
    hs_gemm     <<<dim3(NPAD / 64, BATCH / 128), 256, 0, stream>>>(xHi, xLo, wHi, wLo,
                                                                   bias, logits);
    hs_out      <<<(BATCH * OUTC + 1 + 255) / 256, 256, 0, stream>>>(logits, out);
    hs_penalty  <<<(BATCH * 111) / 256, 256, 0, stream>>>(logits, labels,
                                                          out + (size_t)BATCH * OUTC);
}
